// MoE_44255343018955
// MI455X (gfx1250) — matrix operand order repaired and checked by reading, not compiled, not run on hardware
//
#include <hip/hip_runtime.h>
#include <hip/hip_bf16.h>

#define N_TOK 16384
#define DIM   2048
#define NEXP  8

#define BM 128
#define BN 128
#define BK 32
#define LDT 40   // padded LDS row stride in shorts (80B = 20 banks -> conflict-free)

typedef __attribute__((ext_vector_type(16))) __bf16 v16bf;
typedef __attribute__((ext_vector_type(8)))  float  v8f;

union Frag { uint4 q[2]; v16bf v; };

__device__ __forceinline__ unsigned short f2bf(float f) {
    unsigned u = __float_as_uint(f);
    u += 0x7fffu + ((u >> 16) & 1u);          // round-to-nearest-even
    return (unsigned short)(u >> 16);
}

// Async copy of 16 bytes from global to LDS (CDNA5 ASYNCcnt path, bypasses VGPRs).
// lds_off: LDS byte address (low 32 bits of generic shared pointer).
__device__ __forceinline__ void async_cp16(unsigned lds_off, const void* gptr) {
    asm volatile("global_load_async_to_lds_b128 %0, %1, off"
                 :: "v"(lds_off), "v"((unsigned long long)(uintptr_t)gptr)
                 : "memory");
}

__device__ __forceinline__ void wait_async0() {
    asm volatile("s_wait_asynccnt 0x0" ::: "memory");
}

// ---------------------------------------------------------------------------
// Kernel 1: gate for token 0 only. 8 waves, one expert per wave.
// Writes [w0, w1, (int)i0, (int)i1] to gate_out.
// ---------------------------------------------------------------------------
__global__ void gate_kernel(const float* __restrict__ x,
                            const float* __restrict__ Wg,
                            const float* __restrict__ bg,
                            float* __restrict__ gate_out) {
    __shared__ float logits[NEXP];
    const int w    = threadIdx.x >> 5;   // expert id (8 waves of 32)
    const int lane = threadIdx.x & 31;
    const float* wr = Wg + w * DIM;
    float s = 0.f;
    for (int d = lane; d < DIM; d += 32) s += x[d] * wr[d];
    for (int off = 16; off > 0; off >>= 1) s += __shfl_down(s, off, 32);
    if (lane == 0) logits[w] = s + bg[w];
    __syncthreads();
    if (threadIdx.x == 0) {
        float m = logits[0];
        for (int e = 1; e < NEXP; ++e) m = fmaxf(m, logits[e]);
        float p[NEXP]; float sum = 0.f;
        for (int e = 0; e < NEXP; ++e) { p[e] = __expf(logits[e] - m); sum += p[e]; }
        float inv = 1.f / sum;
        int i0 = 0;
        for (int e = 1; e < NEXP; ++e) if (p[e] > p[i0]) i0 = e;
        int i1 = (i0 == 0) ? 1 : 0;
        for (int e = 0; e < NEXP; ++e) if (e != i0 && p[e] > p[i1]) i1 = e;
        gate_out[0] = p[i0] * inv;
        gate_out[1] = p[i1] * inv;
        ((int*)gate_out)[2] = i0;
        ((int*)gate_out)[3] = i1;
    }
}

// ---------------------------------------------------------------------------
// Kernel 2: x fp32 -> bf16 (8 elems / thread)
// ---------------------------------------------------------------------------
__global__ void cvt_x_kernel(const float* __restrict__ x,
                             unsigned short* __restrict__ xb) {
    long long base = ((long long)blockIdx.x * blockDim.x + threadIdx.x) * 8;
    float4 f0 = *(const float4*)(x + base);
    float4 f1 = *(const float4*)(x + base + 4);
    uint4 o;
    o.x = (unsigned)f2bf(f0.x) | ((unsigned)f2bf(f0.y) << 16);
    o.y = (unsigned)f2bf(f0.z) | ((unsigned)f2bf(f0.w) << 16);
    o.z = (unsigned)f2bf(f1.x) | ((unsigned)f2bf(f1.y) << 16);
    o.w = (unsigned)f2bf(f1.z) | ((unsigned)f2bf(f1.w) << 16);
    *(uint4*)(xb + base) = o;
}

// ---------------------------------------------------------------------------
// Kernel 3: Wc = w0*We[i0] + w1*We[i1] (bf16), bc = w0*be[i0] + w1*be[i1]
// ---------------------------------------------------------------------------
__global__ void combine_kernel(const float* __restrict__ We,
                               const float* __restrict__ be,
                               const float* __restrict__ gate,
                               unsigned short* __restrict__ Wc,
                               float* __restrict__ bc) {
    const float w0 = gate[0], w1 = gate[1];
    const int i0 = ((const int*)gate)[2];
    const int i1 = ((const int*)gate)[3];
    const float* A0 = We + (long long)i0 * DIM * DIM;
    const float* A1 = We + (long long)i1 * DIM * DIM;
    long long base = ((long long)blockIdx.x * blockDim.x + threadIdx.x) * 8;
    float4 a0 = *(const float4*)(A0 + base);
    float4 a1 = *(const float4*)(A1 + base);
    float4 c0 = *(const float4*)(A0 + base + 4);
    float4 c1 = *(const float4*)(A1 + base + 4);
    uint4 o;
    o.x = (unsigned)f2bf(w0*a0.x + w1*a1.x) | ((unsigned)f2bf(w0*a0.y + w1*a1.y) << 16);
    o.y = (unsigned)f2bf(w0*a0.z + w1*a1.z) | ((unsigned)f2bf(w0*a0.w + w1*a1.w) << 16);
    o.z = (unsigned)f2bf(w0*c0.x + w1*c1.x) | ((unsigned)f2bf(w0*c0.y + w1*c1.y) << 16);
    o.w = (unsigned)f2bf(w0*c0.z + w1*c1.z) | ((unsigned)f2bf(w0*c0.w + w1*c1.w) << 16);
    *(uint4*)(Wc + base) = o;

    int gid = blockIdx.x * blockDim.x + threadIdx.x;
    if (gid < DIM / 8) {
        int db = gid * 8;
        for (int t = 0; t < 8; ++t)
            bc[db + t] = w0 * be[i0 * DIM + db + t] + w1 * be[i1 * DIM + db + t];
    }
}

// ---------------------------------------------------------------------------
// Kernel 4: out[N,D] = x_bf16 @ Wc^T + bc  (NT GEMM, K contiguous both sides)
// 256 threads = 8 wave32, block tile 128x128, BK=32, double-buffered LDS
// filled by GLOBAL_LOAD_ASYNC_TO_LDS_B128 (ASYNCcnt). One barrier / k-step.
// Per wave: 2x4 grid of 16x16 WMMA tiles (v_wmma_f32_16x16x32_bf16).
// ---------------------------------------------------------------------------
__global__ __launch_bounds__(256)
void moe_gemm_kernel(const unsigned short* __restrict__ X,
                     const unsigned short* __restrict__ W,
                     const float* __restrict__ bias,
                     float* __restrict__ out) {
    __shared__ unsigned short As[2][BM * LDT];
    __shared__ unsigned short Bs[2][BN * LDT];

    const int tid   = threadIdx.x;
    const int lane  = tid & 31;
    const int wave  = tid >> 5;          // 0..7
    const int lm    = lane & 15;
    const int hi    = lane >> 4;         // 0 or 1

    const int waveM = wave >> 1;         // 0..3 -> 32 rows each
    const int waveN = wave & 1;          // 0..1 -> 64 cols each

    const int blockM = blockIdx.y * BM;
    const int blockN = blockIdx.x * BN;

    // staging assignment: thread t owns 16 shorts (32B) of one row slice
    const int ldRow = tid >> 1;          // 0..127
    const int ldSeg = (tid & 1) * 16;    // short offset within the 32-short slice

    const unsigned short* ag  = X + (long long)(blockM + ldRow) * DIM + ldSeg;
    const unsigned short* bgp = W + (long long)(blockN + ldRow) * DIM + ldSeg;

    unsigned aoff[2], boff[2];
    for (int b = 0; b < 2; ++b) {
        aoff[b] = (unsigned)(uintptr_t)&As[b][ldRow * LDT + ldSeg];
        boff[b] = (unsigned)(uintptr_t)&Bs[b][ldRow * LDT + ldSeg];
    }

    v8f acc[2][4];
    {
        v8f z;
        for (int r = 0; r < 8; ++r) z[r] = 0.f;
        for (int i = 0; i < 2; ++i)
            for (int j = 0; j < 4; ++j)
                acc[i][j] = z;
    }

    float bv[4];
    for (int j = 0; j < 4; ++j)
        bv[j] = bias[blockN + waveN * 64 + j * 16 + lm];

    // prologue: issue async fill of buffer 0 (slab k=0): 4 x 16B per thread
    async_cp16(aoff[0],      ag);
    async_cp16(aoff[0] + 16, ag + 8);
    async_cp16(boff[0],      bgp);
    async_cp16(boff[0] + 16, bgp + 8);

    const int NK = DIM / BK;             // 64
    for (int i = 0; i < NK; ++i) {
        const int cur = i & 1;

        wait_async0();       // this wave's async fills of buf[cur] complete
        __syncthreads();     // all waves filled buf[cur]; all prior reads of it done

        // ---- fragment loads per ISA layout ----
        // A (16x32 bf16): lanes 0-15 hold K {0..7,16..23}, lanes 16-31 {8..15,24..31}
        Frag af[2];
        for (int ii = 0; ii < 2; ++ii) {
            const unsigned short* p = &As[cur][(waveM * 32 + ii * 16 + lm) * LDT] + hi * 8;
            af[ii].q[0] = *(const uint4*)(p);
            af[ii].q[1] = *(const uint4*)(p + 16);
        }
        // B (32x16 bf16): lanes 0-15 hold K {0..7,16..23}, lanes 16-31 {8..15,24..31}
        Frag bf[4];
        for (int jj = 0; jj < 4; ++jj) {
            const unsigned short* p = &Bs[cur][(waveN * 64 + jj * 16 + lm) * LDT] + hi * 8;
            bf[jj].q[0] = *(const uint4*)(p);
            bf[jj].q[1] = *(const uint4*)(p + 16);
        }

        // ---- issue next slab's async fill into the other buffer (overlaps WMMAs)
        if (i + 1 < NK) {
            const int k  = (i + 1) * BK;
            const int nb = cur ^ 1;
            async_cp16(aoff[nb],      ag + k);
            async_cp16(aoff[nb] + 16, ag + k + 8);
            async_cp16(boff[nb],      bgp + k);
            async_cp16(boff[nb] + 16, bgp + k + 8);
        }

        // ---- 8 WMMAs ----
        for (int ii = 0; ii < 2; ++ii)
            for (int jj = 0; jj < 4; ++jj)
                acc[ii][jj] = __builtin_amdgcn_wmma_f32_16x16x32_bf16(
                    false, af[ii].v, false, bf[jj].v,
                    (short)0, acc[ii][jj], false, false);
    }

    // ---- epilogue: C layout = VGPR r -> row r + hi*8, col lm ----
    for (int i = 0; i < 2; ++i) {
        const int rowBase = blockM + waveM * 32 + i * 16 + hi * 8;
        for (int j = 0; j < 4; ++j) {
            const int col = blockN + waveN * 64 + j * 16 + lm;
            const float b = bv[j];
            for (int r = 0; r < 8; ++r)
                out[(long long)(rowBase + r) * DIM + col] = acc[i][j][r] + b;
        }
    }
}

// ---------------------------------------------------------------------------
extern "C" void kernel_launch(void* const* d_in, const int* in_sizes, int n_in,
                              void* d_out, int out_size, void* d_ws, size_t ws_size,
                              hipStream_t stream) {
    (void)in_sizes; (void)n_in; (void)out_size; (void)ws_size;

    const float* x  = (const float*)d_in[0];
    const float* Wg = (const float*)d_in[1];
    const float* bg = (const float*)d_in[2];
    const float* We = (const float*)d_in[3];
    const float* be = (const float*)d_in[4];
    float* out = (float*)d_out;

    char* ws = (char*)d_ws;
    float*          gate = (float*)ws;                           // 16 B
    float*          bc   = (float*)(ws + 256);                   // 8 KB
    unsigned short* Wc   = (unsigned short*)(ws + 256 + DIM * sizeof(float));      // 8 MB
    unsigned short* xb   = (unsigned short*)(ws + 256 + DIM * sizeof(float)
                                             + (size_t)DIM * DIM * sizeof(unsigned short)); // 64 MB

    gate_kernel<<<1, 256, 0, stream>>>(x, Wg, bg, gate);

    cvt_x_kernel<<<(N_TOK * (DIM / 8)) / 256, 256, 0, stream>>>(x, xb);

    combine_kernel<<<(DIM * (DIM / 8)) / 256, 256, 0, stream>>>(We, be, gate, Wc, bc);

    dim3 grid(DIM / BN, N_TOK / BM);   // (16, 128)
    moe_gemm_kernel<<<grid, 256, 0, stream>>>(xb, Wc, bc, out);
}
